// BasicBlock_deformable_attention_6433861009840
// MI455X (gfx1250) — matrix operand order repaired and checked by reading, not compiled, not run on hardware
//
#include <hip/hip_runtime.h>
#include <hip/hip_bf16.h>

typedef _Float16 f16;
typedef _Float16 v16h __attribute__((ext_vector_type(16)));
typedef _Float16 v8h  __attribute__((ext_vector_type(8)));
typedef float    v8f  __attribute__((ext_vector_type(8)));

static constexpr int Bsz = 64, Himg = 64, Wimg = 64, HW = Himg * Wimg;
static constexpr float BN_EPS = 1e-5f;

// ---------------------------------------------------------------------------
// Implicit-GEMM conv via V_WMMA_F32_16X16X32_F16.
// Block: 256 threads (8 wave32), covers ROWS*64 pixels of one image x COUT_BLK
// output channels. GEMM: M = pixels, N = out channels, K = KS*KS*CIN
// (tap-major, channel-minor). A staged in LDS as [row][col][c] (f16),
// B (weights) as [tap][o][c] (f16) -> all fragment runs are contiguous b128
// LDS loads. f32 accumulate; raw (pre-BN) output + optional bias.
// ---------------------------------------------------------------------------
template <int CIN, int COUT, int COUT_BLK, int KS, int ROWS>
__global__ __launch_bounds__(256) void conv_wmma(
    const float* __restrict__ in, const float* __restrict__ wgt,
    const float* __restrict__ bias, float* __restrict__ out) {
  static_assert(KS == 1 || ROWS == 1, "halo staging assumes ROWS==1 for 3x3");
  constexpr int TAPS   = KS * KS;
  constexpr int K_TOT  = TAPS * CIN;
  constexpr int KSTEPS = (K_TOT + 31) / 32;
  constexpr int W_LDS  = (KS == 3) ? (Wimg + 2) : Wimg;
  constexpr int R_LDS  = (KS == 3) ? 3 : ROWS;
  constexpr int MPIX   = ROWS * Wimg;
  constexpr int MT     = MPIX / 16;
  constexpr int NT     = COUT_BLK / 16;

  __shared__ f16 lds_in[R_LDS * W_LDS * CIN];
  __shared__ f16 lds_w[TAPS * COUT_BLK * CIN];

  const int tid  = threadIdx.x;
  const int lane = tid & 31;
  const int wave = tid >> 5;
  const int blocks_per_img = Himg / ROWS;
  const int b         = blockIdx.x / blocks_per_img;
  const int y_base    = (blockIdx.x % blocks_per_img) * ROWS;
  const int cout_base = blockIdx.y * COUT_BLK;

  // ---- stage input tile (with halo + zero padding) as f16, channel-minor ----
  for (int pr = tid; pr < R_LDS * W_LDS; pr += 256) {
    const int r   = pr / W_LDS;
    const int col = pr % W_LDS;
    const int gy  = (KS == 3) ? (y_base + r - 1) : (y_base + r);
    const int gx  = (KS == 3) ? (col - 1) : col;
    const bool ok = (gy >= 0) && (gy < Himg) && (gx >= 0) && (gx < Wimg);
    for (int c = 0; c < CIN; ++c) {
      float v = ok ? in[(((size_t)b * CIN + c) * Himg + gy) * Wimg + gx] : 0.f;
      lds_in[(r * W_LDS + col) * CIN + c] = (f16)v;
    }
  }
  // ---- stage weights [tap][o_local][c], zero-pad o >= COUT ----
  for (int idx = tid; idx < TAPS * COUT_BLK * CIN; idx += 256) {
    const int tap = idx / (COUT_BLK * CIN);
    const int rem = idx % (COUT_BLK * CIN);
    const int ol  = rem / CIN;
    const int c   = rem % CIN;
    const int o   = cout_base + ol;
    float v = (o < COUT) ? wgt[((size_t)o * CIN + c) * TAPS + tap] : 0.f;
    lds_w[idx] = (f16)v;
  }
  __syncthreads();

  for (int t = wave; t < MT * NT; t += 8) {
    const int mt = t % MT;
    const int nt = t / MT;
    const int m  = lane & 15;
    const int hi = lane >> 4;
    const int p  = mt * 16 + m;
    const int prow = p / Wimg;
    const int px   = p % Wimg;

    v8f acc = {0.f, 0.f, 0.f, 0.f, 0.f, 0.f, 0.f, 0.f};

#pragma unroll
    for (int kk = 0; kk < KSTEPS; ++kk) {
      v16h a, bf;
      // A fragment: lane holds K = kk*32 + hi*8 + {0..7} and +16..+23
#pragma unroll
      for (int run = 0; run < 2; ++run) {
        const int k0 = kk * 32 + hi * 8 + run * 16;
        v8h av;
        if (k0 < K_TOT) {
          const int tap = k0 / CIN, c0 = k0 % CIN;
          const int dy = tap / KS, dx = tap % KS;
          const int rr = (KS == 3) ? dy : prow;
          const int cc = (KS == 3) ? (px + dx) : px;
          av = *(const v8h*)&lds_in[(rr * W_LDS + cc) * CIN + c0];
        } else {
#pragma unroll
          for (int e = 0; e < 8; ++e) av[e] = (f16)0.f;
        }
        ((v8h*)&a)[run] = av;
      }
      // B fragment: lane col n = lane&15 holds K = kk*32 + hi*8 + {0..7} and +16..+23
      {
        const int ol = nt * 16 + (lane & 15);
#pragma unroll
        for (int run = 0; run < 2; ++run) {
          const int kr = kk * 32 + hi * 8 + run * 16;
          v8h bv;
          if (kr < K_TOT) {
            const int tap = kr / CIN, c0 = kr % CIN;
            bv = *(const v8h*)&lds_w[(tap * COUT_BLK + ol) * CIN + c0];
          } else {
#pragma unroll
            for (int e = 0; e < 8; ++e) bv[e] = (f16)0.f;
          }
          ((v8h*)&bf)[run] = bv;
        }
      }
      acc = __builtin_amdgcn_wmma_f32_16x16x32_f16(
          false, a, false, bf, (short)0, acc, false, false);
    }

    // D layout: VGPR v -> M = v + hi*8, N = lane&15 => 8 consecutive x per lane
    const int o = cout_base + nt * 16 + (lane & 15);
    if (o < COUT) {
      const float bv = bias ? bias[o] : 0.f;
      const int p0   = mt * 16 + hi * 8;
      const int row0 = p0 / Wimg;
      const int x0   = p0 % Wimg;
      float* dst = out + (((size_t)b * COUT + o) * Himg + (y_base + row0)) * Wimg + x0;
#pragma unroll
      for (int v = 0; v < 8; ++v) dst[v] = acc[v] + bv;
    }
  }
}

// ---------------------------------------------------------------------------
// Per-channel sum / sum-of-squares for training-mode BN (over N,H,W).
// grid = (C, 16); block-level LDS reduction then one atomicAdd pair per block.
// ---------------------------------------------------------------------------
__global__ __launch_bounds__(256) void stats_kernel(
    const float* __restrict__ t, float* __restrict__ st, int C) {
  const int c     = blockIdx.x;
  const int total = Bsz * HW;
  const int per   = total / gridDim.y;
  const int start = blockIdx.y * per;
  float s = 0.f, q = 0.f;
  for (int j = start + threadIdx.x; j < start + per; j += 256) {
    const int bb = j / HW, i = j % HW;
    const float v = t[((size_t)bb * C + c) * HW + i];
    s += v;
    q += v * v;
  }
  __shared__ float rs[256], rq[256];
  rs[threadIdx.x] = s;
  rq[threadIdx.x] = q;
  __syncthreads();
  for (int off = 128; off > 0; off >>= 1) {
    if (threadIdx.x < off) {
      rs[threadIdx.x] += rs[threadIdx.x + off];
      rq[threadIdx.x] += rq[threadIdx.x + off];
    }
    __syncthreads();
  }
  if (threadIdx.x == 0) {
    atomicAdd(&st[2 * c], rs[0]);
    atomicAdd(&st[2 * c + 1], rq[0]);
  }
}

// ---------------------------------------------------------------------------
// BN apply (+ optional ReLU / residual add).
// ---------------------------------------------------------------------------
template <bool RELU, bool RESID>
__global__ __launch_bounds__(256) void bn_apply(
    const float* __restrict__ t, const float* __restrict__ st,
    const float* __restrict__ g, const float* __restrict__ be,
    const float* __restrict__ res, float* __restrict__ out, int C) {
  const float invN = 1.0f / (float)(Bsz * HW);
  const size_t n = (size_t)Bsz * C * HW;
  for (size_t idx = (size_t)blockIdx.x * 256 + threadIdx.x; idx < n;
       idx += (size_t)gridDim.x * 256) {
    const int c = (int)((idx / HW) % C);
    const float mean = st[2 * c] * invN;
    const float var  = st[2 * c + 1] * invN - mean * mean;
    float v = (t[idx] - mean) * rsqrtf(var + BN_EPS) * g[c] + be[c];
    if (RESID) v += res[idx];
    if (RELU) v = fmaxf(v, 0.f);
    out[idx] = v;
  }
}

// ---------------------------------------------------------------------------
// Deformable 3x3 conv (C=16 -> O=16): bilinear gather (L2-resident a2) +
// small dense MAC per tap. Weights broadcast from LDS in [k][o][c] order.
// Tap loop deliberately NOT unrolled: keeps per-tap sampling state (~40 VGPRs)
// from multiplying by 9 (previous version hit 512 VGPRs + 8.7KB spill).
// ---------------------------------------------------------------------------
__global__ __launch_bounds__(256) void deform_kernel(
    const float* __restrict__ a2, const float* __restrict__ offs,
    const float* __restrict__ w3, float* __restrict__ out) {
  __shared__ float wsm[9 * 16 * 16];  // [k][o][c]
  for (int i = threadIdx.x; i < 16 * 16 * 9; i += 256) {
    const int o = i / (16 * 9);
    const int c = (i / 9) % 16;
    const int k = i % 9;
    wsm[(k * 16 + o) * 16 + c] = w3[i];  // w3 is [o][c][k]
  }
  __syncthreads();

  const int pid = blockIdx.x * 256 + threadIdx.x;
  if (pid >= Bsz * HW) return;
  const int b = pid / HW;
  const int i = pid % HW;
  const int y = i / Wimg, x = i % Wimg;
  const float* offp = offs + (size_t)b * 18 * HW + i;
  const float* xb   = a2 + (size_t)b * 16 * HW;

  float acc[16];
#pragma unroll
  for (int o = 0; o < 16; ++o) acc[o] = 0.f;

#pragma unroll 1
  for (int k = 0; k < 9; ++k) {
    const int ky = k / 3, kx = k % 3;
    // padded coords: p0 = (y+1, x+1), tap offset (ky-1, kx-1)
    float py = (float)(y + ky) + offp[(size_t)k * HW];
    float px = (float)(x + kx) + offp[(size_t)(9 + k) * HW];
    py = fminf(fmaxf(py, 0.f), 65.f);
    px = fminf(fmaxf(px, 0.f), 65.f);
    const int y0 = (int)floorf(py), x0 = (int)floorf(px);
    const int y1 = min(y0 + 1, 65), x1 = min(x0 + 1, 65);
    const float wy = py - (float)y0, wx = px - (float)x0;
    const float w00 = (1.f - wy) * (1.f - wx), w01 = (1.f - wy) * wx;
    const float w10 = wy * (1.f - wx), w11 = wy * wx;
    // padded index v is real data iff 1 <= v <= 64 (border row/col is zero pad)
    const bool vy0 = (y0 >= 1) && (y0 <= 64), vy1 = (y1 >= 1) && (y1 <= 64);
    const bool vx0 = (x0 >= 1) && (x0 <= 64), vx1 = (x1 >= 1) && (x1 <= 64);
    const int iy0 = (y0 - 1) * Wimg, iy1 = (y1 - 1) * Wimg;
    const int ix0 = x0 - 1, ix1 = x1 - 1;

    float s[16];
#pragma unroll
    for (int c = 0; c < 16; ++c) {
      const float* ch = xb + (size_t)c * HW;
      const float v00 = (vy0 && vx0) ? ch[iy0 + ix0] : 0.f;
      const float v01 = (vy0 && vx1) ? ch[iy0 + ix1] : 0.f;
      const float v10 = (vy1 && vx0) ? ch[iy1 + ix0] : 0.f;
      const float v11 = (vy1 && vx1) ? ch[iy1 + ix1] : 0.f;
      s[c] = v00 * w00 + v01 * w01 + v10 * w10 + v11 * w11;
    }
    const float* wk = &wsm[k * 256];
#pragma unroll
    for (int o = 0; o < 16; ++o) {
      float a = 0.f;
#pragma unroll
      for (int c = 0; c < 16; ++c) a += wk[o * 16 + c] * s[c];
      acc[o] += a;
    }
  }
  float* op = out + (size_t)b * 16 * HW + i;
#pragma unroll
  for (int o = 0; o < 16; ++o) op[(size_t)o * HW] = acc[o];
}

// ---------------------------------------------------------------------------
extern "C" void kernel_launch(void* const* d_in, const int* in_sizes, int n_in,
                              void* d_out, int out_size, void* d_ws,
                              size_t ws_size, hipStream_t stream) {
  (void)in_sizes; (void)n_in; (void)out_size; (void)ws_size;
  const float* x  = (const float*)d_in[0];
  const float* w1 = (const float*)d_in[1];
  const float* g1 = (const float*)d_in[2];
  const float* b1 = (const float*)d_in[3];
  const float* w2 = (const float*)d_in[4];
  const float* g2 = (const float*)d_in[5];
  const float* b2 = (const float*)d_in[6];
  const float* wo = (const float*)d_in[7];
  const float* bo = (const float*)d_in[8];
  const float* w3 = (const float*)d_in[9];
  const float* g3 = (const float*)d_in[10];
  const float* b3 = (const float*)d_in[11];
  const float* w4 = (const float*)d_in[12];
  const float* g4 = (const float*)d_in[13];
  const float* b4 = (const float*)d_in[14];
  float* out = (float*)d_out;

  char* ws = (char*)d_ws;
  const size_t sz64 = (size_t)Bsz * 64 * HW * sizeof(float);
  const size_t sz16 = (size_t)Bsz * 16 * HW * sizeof(float);
  const size_t sz18 = (size_t)Bsz * 18 * HW * sizeof(float);
  float* R0 = (float*)ws; ws += sz64;   // t1, later t4
  float* R1 = (float*)ws; ws += sz64;   // a1, later a3
  float* R2 = (float*)ws; ws += sz16;   // t2, later t3
  float* R3 = (float*)ws; ws += sz16;   // a2
  float* R4 = (float*)ws; ws += sz18;   // offsets
  float* ST = (float*)ws; ws += 4 * 128 * sizeof(float);
  float* S0 = ST, *S1 = ST + 128, *S2 = ST + 256, *S3 = ST + 384;

  hipMemsetAsync(ST, 0, 4 * 128 * sizeof(float), stream);

  // conv1 3x3 64->64 (Cout split across grid.y to fit LDS)
  conv_wmma<64, 64, 32, 3, 1><<<dim3(Bsz * Himg, 2), 256, 0, stream>>>(x, w1, nullptr, R0);
  stats_kernel<<<dim3(64, 16), 256, 0, stream>>>(R0, S0, 64);
  bn_apply<true, false><<<2048, 256, 0, stream>>>(R0, S0, g1, b1, nullptr, R1, 64);

  // conv2 1x1 64->16 (2 rows per block so all 8 waves have a tile)
  conv_wmma<64, 16, 16, 1, 2><<<dim3(Bsz * Himg / 2, 1), 256, 0, stream>>>(R1, w2, nullptr, R2);
  stats_kernel<<<dim3(16, 16), 256, 0, stream>>>(R2, S1, 16);
  bn_apply<true, false><<<1024, 256, 0, stream>>>(R2, S1, g2, b2, nullptr, R3, 16);

  // offset conv 3x3 16->18 (+bias, no BN); N padded to 32, stores masked
  conv_wmma<16, 18, 32, 3, 1><<<dim3(Bsz * Himg, 1), 256, 0, stream>>>(R3, wo, bo, R4);

  // deformable conv 3x3 16->16 (bilinear gather + dense MAC)
  deform_kernel<<<(Bsz * HW) / 256, 256, 0, stream>>>(R3, R4, w3, R2);
  stats_kernel<<<dim3(16, 16), 256, 0, stream>>>(R2, S2, 16);
  bn_apply<true, false><<<1024, 256, 0, stream>>>(R2, S2, g3, b3, nullptr, R1, 16);

  // conv4 3x3 16->64
  conv_wmma<16, 64, 64, 3, 1><<<dim3(Bsz * Himg, 1), 256, 0, stream>>>(R1, w4, nullptr, R0);
  stats_kernel<<<dim3(64, 16), 256, 0, stream>>>(R0, S3, 64);

  // bn4 + residual + relu -> d_out
  bn_apply<true, true><<<2048, 256, 0, stream>>>(R0, S3, g4, b4, x, out, 64);
}
